// GraphNetwork_35003983462587
// MI455X (gfx1250) — matrix operand order repaired and checked by reading, not compiled, not run on hardware
//
#include <hip/hip_runtime.h>
#include <hip/hip_bf16.h>

#define N_NODES 50000
#define N_EDGES 1200000
#define FDIM 64
#define HDIM 64
#define H2DIM 32
#define GDIM 256
#define CDIM 10
#define EPSV 1e-5f

typedef __attribute__((ext_vector_type(16))) __bf16 v16bf;
typedef __attribute__((ext_vector_type(8)))  float v8f;
typedef __attribute__((ext_vector_type(4)))  unsigned int u32x4;

union FragBF { u32x4 u[2]; v16bf v; };

// ---------------- utility kernels ----------------

__global__ void zero_f32(float* __restrict__ p, long n) {
    long i = (long)blockIdx.x * blockDim.x + threadIdx.x;
    if (i < n) p[i] = 0.0f;
}

__global__ void cvt_bf16(const float* __restrict__ in, __bf16* __restrict__ out, long n) {
    long i = (long)blockIdx.x * blockDim.x + threadIdx.x;
    if (i < n) out[i] = (__bf16)in[i];
}

// degree: atomic count of in-edges
__global__ void edge_deg(const int* __restrict__ dst, float* __restrict__ deg) {
    long e = (long)blockIdx.x * blockDim.x + threadIdx.x;
    if (e < N_EDGES) atomicAdd(deg + dst[e], 1.0f);
}

// dinv = rsqrt(deg + 1) in place
__global__ void deg_rsqrt(float* __restrict__ d, long n) {
    long i = (long)blockIdx.x * blockDim.x + threadIdx.x;
    if (i < n) d[i] = rsqrtf(d[i] + 1.0f);
}

// Pack W[64 x Hout] (row-major, k-major) into WMMA bf16 B-fragment layout:
// frag[ntile][khalf][lane][e], k = khalf*32 + (lane>=16?8:0) + (e&7) + (e>=8?16:0), n = ntile*16 + lane%16
__global__ void pack_w(const float* __restrict__ W, __bf16* __restrict__ out, int Hout) {
    int idx = blockIdx.x * blockDim.x + threadIdx.x;
    int ntiles = Hout >> 4;
    int total = ntiles * 2 * 32 * 16;
    if (idx >= total) return;
    int e    = idx & 15;
    int lane = (idx >> 4) & 31;
    int kh   = (idx >> 9) & 1;
    int nt   = idx >> 10;
    int k = kh * 32 + ((lane >= 16) ? 8 : 0) + (e & 7) + ((e >> 3) << 4);
    int n = nt * 16 + (lane & 15);
    out[idx] = (__bf16)W[k * Hout + n];
}

// ---------------- WMMA GEMM:  T[N x Hout] = X[N x 64] @ W  ----------------
// block = 128 threads = 4 waves; wave w -> 16-col tile w; grid.x = N/16 (exact)
__global__ void __launch_bounds__(128)
gcn_gemm_wmma(const __bf16* __restrict__ X, const __bf16* __restrict__ Wf,
              float* __restrict__ T, int Hout) {
    const int lane = threadIdx.x & 31;
    const int wv   = threadIdx.x >> 5;
    const int ntiles = Hout >> 4;
    if (wv >= ntiles) return;                 // wave-uniform; EXEC all-1s inside
    const int row0  = blockIdx.x << 4;
    const int lhalf = (lane >> 4) & 1;

    // A fragments (16x32 bf16 each), per ISA 16-bit A layout:
    // lane's 16 elems = two contiguous 8-elem runs at k = base, base+16
    const __bf16* pa = X + (size_t)(row0 + (lane & 15)) * FDIM + lhalf * 8;
    FragBF a0, a1, b0, b1;
    a0.u[0] = *(const u32x4*)(pa +  0);
    a0.u[1] = *(const u32x4*)(pa + 16);
    a1.u[0] = *(const u32x4*)(pa + 32);
    a1.u[1] = *(const u32x4*)(pa + 48);

    // B fragments: pre-packed contiguous 32B per lane
    const __bf16* pb0 = Wf + (size_t)(((wv * 2 + 0) * 32 + lane) * 16);
    const __bf16* pb1 = Wf + (size_t)(((wv * 2 + 1) * 32 + lane) * 16);
    b0.u[0] = ((const u32x4*)pb0)[0];  b0.u[1] = ((const u32x4*)pb0)[1];
    b1.u[0] = ((const u32x4*)pb1)[0];  b1.u[1] = ((const u32x4*)pb1)[1];

    v8f c = {};
    c = __builtin_amdgcn_wmma_f32_16x16x32_bf16(false, a0.v, false, b0.v, (short)0, c, false, false);
    c = __builtin_amdgcn_wmma_f32_16x16x32_bf16(false, a1.v, false, b1.v, (short)0, c, false, false);

    // C/D layout: VGPR r, lanes 0-15 -> M=r, lanes 16-31 -> M=8+r; N = lane%16
    const int col   = (wv << 4) + (lane & 15);
    const int rbase = row0 + lhalf * 8;
    #pragma unroll
    for (int r = 0; r < 8; ++r)
        T[(size_t)(rbase + r) * Hout + col] = c[r];
}

// ---------------- aggregation ----------------

// agg = T * dinv^2 (self-loop) + bias  (initializes agg fully; no pre-zero needed)
__global__ void selfloop_bias(const float* __restrict__ T, const float* __restrict__ dinv,
                              const float* __restrict__ b, float* __restrict__ agg,
                              int H, long total) {
    long i = (long)blockIdx.x * blockDim.x + threadIdx.x;
    if (i >= total) return;
    int node = (int)(i / H);
    int c    = (int)(i % H);
    float d = dinv[node];
    agg[i] = T[i] * d * d + b[c];
}

// agg[dst] += T[src] * (dinv[src]*dinv[dst]); thread = (edge, 4-channel group)
__global__ void edge_scatter(const float* __restrict__ T, const float* __restrict__ dinv,
                             const int* __restrict__ src, const int* __restrict__ dst,
                             float* __restrict__ agg, int cgs, int H) {
    long i = (long)blockIdx.x * blockDim.x + threadIdx.x;
    long total = (long)N_EDGES * cgs;
    if (i >= total) return;
    int e  = (int)(i / cgs);
    int cg = (int)(i % cgs);
    int s = src[e], d = dst[e];
    float nm = dinv[s] * dinv[d];
    const float4 v = *(const float4*)(T + (size_t)s * H + cg * 4);
    float* p = agg + (size_t)d * H + cg * 4;
    atomicAdd(p + 0, v.x * nm);
    atomicAdd(p + 1, v.y * nm);
    atomicAdd(p + 2, v.z * nm);
    atomicAdd(p + 3, v.w * nm);
}

// ---------------- batch norm ----------------

__global__ void bn_stats(const float* __restrict__ A, float* __restrict__ gsum,
                         float* __restrict__ gsq, int H) {
    __shared__ float ssum[64], ssq[64];
    const int c    = threadIdx.x % H;
    const int rpb  = blockDim.x / H;
    const int rloc = threadIdx.x / H;
    if (threadIdx.x < H) { ssum[threadIdx.x] = 0.f; ssq[threadIdx.x] = 0.f; }
    __syncthreads();
    float ls = 0.f, lq = 0.f;
    for (int r = blockIdx.x * rpb + rloc; r < N_NODES; r += gridDim.x * rpb) {
        float v = A[(size_t)r * H + c];
        ls += v; lq += v * v;
    }
    atomicAdd(&ssum[c], ls);
    atomicAdd(&ssq[c], lq);
    __syncthreads();
    if (threadIdx.x < H) {
        atomicAdd(&gsum[threadIdx.x], ssum[threadIdx.x]);
        atomicAdd(&gsq[threadIdx.x],  ssq[threadIdx.x]);
    }
}

__global__ void bn_param(const float* __restrict__ gsum, const float* __restrict__ gsq,
                         const float* __restrict__ gamma, const float* __restrict__ beta,
                         float* __restrict__ sc, float* __restrict__ sh, int H) {
    int c = threadIdx.x;
    if (c >= H) return;
    float mu  = gsum[c] * (1.0f / (float)N_NODES);
    float var = gsq[c]  * (1.0f / (float)N_NODES) - mu * mu;
    float s = gamma[c] * rsqrtf(var + EPSV);
    sc[c] = s;
    sh[c] = beta[c] - mu * s;
}

// x = relu(agg*scale + shift); write bf16 (next GEMM) and/or f32 (pool input)
__global__ void bn_apply(const float* __restrict__ A, const float* __restrict__ sc,
                         const float* __restrict__ sh, __bf16* __restrict__ obf,
                         float* __restrict__ of32, int H, long total) {
    long i = (long)blockIdx.x * blockDim.x + threadIdx.x;
    if (i >= total) return;
    int c = (int)(i % H);
    float v = fmaxf(A[i] * sc[c] + sh[c], 0.0f);
    if (obf)  obf[i]  = (__bf16)v;
    if (of32) of32[i] = v;
}

// ---------------- pool + classifier ----------------

__global__ void pool_sum(const float* __restrict__ hf, const int* __restrict__ batch,
                         float* __restrict__ psum, float* __restrict__ pcnt) {
    long i = (long)blockIdx.x * blockDim.x + threadIdx.x;
    long total = (long)N_NODES * H2DIM;
    if (i >= total) return;
    int node = (int)(i >> 5);
    int c    = (int)(i & 31);
    int g = batch[node];
    atomicAdd(psum + (size_t)g * H2DIM + c, hf[i]);
    if (c == 0) atomicAdd(pcnt + g, 1.0f);
}

__global__ void classify_softmax(const float* __restrict__ psum, const float* __restrict__ pcnt,
                                 const float* __restrict__ Wl, const float* __restrict__ bl,
                                 float* __restrict__ out) {
    int g = blockIdx.x;
    int t = threadIdx.x;
    __shared__ float lg[CDIM];
    __shared__ float invsum;
    float inv = 1.0f / fmaxf(pcnt[g], 1.0f);
    if (t < CDIM) {
        float acc = bl[t];
        #pragma unroll
        for (int k = 0; k < H2DIM; ++k)
            acc += psum[(size_t)g * H2DIM + k] * inv * Wl[k * CDIM + t];
        lg[t] = acc;
    }
    __syncthreads();
    if (t == 0) {
        float m = lg[0];
        for (int j = 1; j < CDIM; ++j) m = fmaxf(m, lg[j]);
        float s = 0.f;
        for (int j = 0; j < CDIM; ++j) { float e = __expf(lg[j] - m); lg[j] = e; s += e; }
        invsum = 1.0f / s;
    }
    __syncthreads();
    if (t < CDIM) out[(size_t)g * CDIM + t] = lg[t] * invsum;
}

// ---------------- launch ----------------

static inline size_t alignup(size_t v) { return (v + 255) & ~(size_t)255; }

extern "C" void kernel_launch(void* const* d_in, const int* in_sizes, int n_in,
                              void* d_out, int out_size, void* d_ws, size_t ws_size,
                              hipStream_t stream) {
    const float* x    = (const float*)d_in[0];
    const int*   esrc = (const int*)d_in[1];
    const int*   edst = (const int*)d_in[2];
    const int*   batch= (const int*)d_in[3];
    const float* Wmat[4] = { (const float*)d_in[4], (const float*)d_in[6],
                             (const float*)d_in[8], (const float*)d_in[10] };
    const float* bvec[4] = { (const float*)d_in[5], (const float*)d_in[7],
                             (const float*)d_in[9], (const float*)d_in[11] };
    const float* gvec[4] = { (const float*)d_in[12], (const float*)d_in[14],
                             (const float*)d_in[16], (const float*)d_in[18] };
    const float* bevec[4]= { (const float*)d_in[13], (const float*)d_in[15],
                             (const float*)d_in[17], (const float*)d_in[19] };
    const float* Wl = (const float*)d_in[20];
    const float* bl = (const float*)d_in[21];
    float* out = (float*)d_out;

    // workspace carve-up
    char* w = (char*)d_ws;
    size_t off = 0;
    auto grab = [&](size_t bytes) { char* p = w + off; off = alignup(off + bytes); return p; };
    float*  dinv  = (float*)grab((size_t)N_NODES * 4);
    float*  tbuf  = (float*)grab((size_t)N_NODES * HDIM * 4);
    float*  agg   = (float*)grab((size_t)N_NODES * HDIM * 4);
    __bf16* xb0   = (__bf16*)grab((size_t)N_NODES * HDIM * 2);
    __bf16* xb1   = (__bf16*)grab((size_t)N_NODES * HDIM * 2);
    __bf16* wf[4];
    for (int l = 0; l < 4; ++l) wf[l] = (__bf16*)grab(4096 * 2);
    float*  gsum  = (float*)grab(64 * 4);
    float*  gsq   = (float*)grab(64 * 4);
    float*  sc    = (float*)grab(64 * 4);
    float*  sh    = (float*)grab(64 * 4);
    float*  psum  = (float*)grab((size_t)GDIM * H2DIM * 4);
    float*  pcnt  = (float*)grab((size_t)GDIM * 4);
    (void)ws_size; (void)n_in; (void)in_sizes; (void)out_size;

    const int TB = 256;
    auto blocks = [&](long n) { return (unsigned)((n + TB - 1) / TB); };

    // degree -> dinv
    zero_f32<<<blocks(N_NODES), TB, 0, stream>>>(dinv, N_NODES);
    edge_deg<<<blocks(N_EDGES), TB, 0, stream>>>(edst, dinv);
    deg_rsqrt<<<blocks(N_NODES), TB, 0, stream>>>(dinv, N_NODES);

    // input -> bf16, weights -> packed bf16 fragments
    cvt_bf16<<<blocks((long)N_NODES * FDIM), TB, 0, stream>>>(x, xb0, (long)N_NODES * FDIM);
    pack_w<<<16, TB, 0, stream>>>(Wmat[0], wf[0], HDIM);
    pack_w<<<16, TB, 0, stream>>>(Wmat[1], wf[1], HDIM);
    pack_w<<<16, TB, 0, stream>>>(Wmat[2], wf[2], HDIM);
    pack_w<<< 8, TB, 0, stream>>>(Wmat[3], wf[3], H2DIM);

    __bf16* xin = xb0;
    __bf16* xout = xb1;
    for (int l = 0; l < 4; ++l) {
        const int Hout = (l == 3) ? H2DIM : HDIM;
        const long tot = (long)N_NODES * Hout;
        const int  cgs = Hout / 4;

        gcn_gemm_wmma<<<N_NODES / 16, 128, 0, stream>>>(xin, wf[l], tbuf, Hout);
        selfloop_bias<<<blocks(tot), TB, 0, stream>>>(tbuf, dinv, bvec[l], agg, Hout, tot);
        edge_scatter<<<blocks((long)N_EDGES * cgs), TB, 0, stream>>>(tbuf, dinv, esrc, edst,
                                                                     agg, cgs, Hout);
        zero_f32<<<1, 128, 0, stream>>>(gsum, 64);
        zero_f32<<<1, 128, 0, stream>>>(gsq, 64);
        bn_stats<<<256, TB, 0, stream>>>(agg, gsum, gsq, Hout);
        bn_param<<<1, 64, 0, stream>>>(gsum, gsq, gvec[l], bevec[l], sc, sh, Hout);
        if (l < 3) {
            bn_apply<<<blocks(tot), TB, 0, stream>>>(agg, sc, sh, xout, nullptr, Hout, tot);
        } else {
            // final layer: fp32 activations (into tbuf, free by now) for pooling
            bn_apply<<<blocks(tot), TB, 0, stream>>>(agg, sc, sh, nullptr, tbuf, Hout, tot);
        }
        __bf16* tmp = xin; xin = xout; xout = tmp;
    }

    // pool + classifier + softmax
    zero_f32<<<blocks((long)GDIM * H2DIM), TB, 0, stream>>>(psum, (long)GDIM * H2DIM);
    zero_f32<<<1, GDIM, 0, stream>>>(pcnt, GDIM);
    pool_sum<<<blocks((long)N_NODES * H2DIM), TB, 0, stream>>>(tbuf, batch, psum, pcnt);
    classify_softmax<<<GDIM, 32, 0, stream>>>(psum, pcnt, Wl, bl, out);
}
